// GAT_linear_negbin_29832842838722
// MI455X (gfx1250) — matrix operand order repaired and checked by reading, not compiled, not run on hardware
//
#include <hip/hip_runtime.h>
#include <hip/hip_bf16.h>

// ---------------------------------------------------------------------------
// GAT forward (2 layers + variance head) for gfx1250.
// GEMMs: v_wmma_f32_16x16x32_bf16, B pre-packed into WMMA fragment layout,
// one wave computes a 16 x NC row-strip; full-tile fast path avoids EXEC
// manipulation and WMMA->VALU hazard NOPs in the hot loop.
// Edge softmax / scatter: fp32 atomics; self-loops via node kernels.
// ---------------------------------------------------------------------------

typedef __attribute__((ext_vector_type(16))) __bf16          v16bf;
typedef __attribute__((ext_vector_type(8)))  float           v8f;
typedef __attribute__((ext_vector_type(8)))  unsigned short  v8us;
typedef __attribute__((ext_vector_type(16))) unsigned short  v16us;

#define LEAKY_SLOPE 0.2f
#define TPB 256

__device__ inline unsigned short f2bf(float f) {
  unsigned int u = __float_as_uint(f);
  u += 0x7FFFu + ((u >> 16) & 1u);          // round-to-nearest-even
  return (unsigned short)(u >> 16);
}

__device__ inline void atomicMaxF(float* addr, float val) {
  unsigned int* ua = (unsigned int*)addr;
  unsigned int old = *ua;
  while (true) {
    float cur = __uint_as_float(old);
    if (cur >= val) break;
    unsigned int assumed = old;
    old = atomicCAS(ua, assumed, __float_as_uint(val));
    if (old == assumed) break;
  }
}

__device__ __forceinline__ v16bf load_a_frag(const unsigned short* p) {
  v8us lo = *(const v8us*)p;                // k offsets 0..7 of this lane half
  v8us hv = *(const v8us*)(p + 16);         // k offsets 16..23 of this lane half
  v16us au = __builtin_shufflevector(lo, hv, 0, 1, 2, 3, 4, 5, 6, 7,
                                              8, 9, 10, 11, 12, 13, 14, 15);
  return __builtin_bit_cast(v16bf, au);
}

// ------------------------------ utility kernels ----------------------------

__global__ void k_cvt_bf16(const float* __restrict__ s, unsigned short* __restrict__ d, long n) {
  long i = (long)blockIdx.x * blockDim.x + threadIdx.x;
  long stride = (long)gridDim.x * blockDim.x;
  for (; i < n; i += stride) d[i] = f2bf(s[i]);
}

__global__ void k_init_f32(float* __restrict__ p, float v, long n) {
  long i = (long)blockIdx.x * blockDim.x + threadIdx.x;
  long stride = (long)gridDim.x * blockDim.x;
  for (; i < n; i += stride) p[i] = v;
}

// Pack W[K,NC] (f32, row-major) into WMMA B-fragment order (bf16):
// Bp[(kt*Ntiles + nt)*512 + lane*16 + e] = bf16(W[k,n])
//   k = kt*32 + (e>=8 ? 16 : 0) + (lane>=16 ? 8 : 0) + (e&7),  n = nt*16 + (lane&15)
__global__ void k_pack_b(const float* __restrict__ W, unsigned short* __restrict__ Bp,
                         int K, int NC) {
  const int Ntiles = NC >> 4;
  const long total = (long)(K >> 5) * Ntiles * 512;
  long i = (long)blockIdx.x * blockDim.x + threadIdx.x;
  if (i >= total) return;
  int e    = (int)(i & 15);
  int lane = (int)((i >> 4) & 31);
  long ft  = i >> 9;                         // kt*Ntiles + nt
  int nt = (int)(ft % Ntiles);
  int kt = (int)(ft / Ntiles);
  int k = (kt << 5) + ((e >> 3) << 4) + ((lane >> 4) << 3) + (e & 7);
  int n = (nt << 4) + (lane & 15);
  Bp[i] = f2bf(W[(size_t)k * NC + n]);
}

// ------------------------------ WMMA GEMM ----------------------------------
// C[M, NTILES*16] = A[M,K](bf16 row-major) * Bp(packed fragments) (+ bias).
// One wave computes a 16 x (NTILES*16) strip; A fragment reused per k-step.
template <int NTILES>
__global__ void k_gemm_wmma(const unsigned short* __restrict__ A,
                            const unsigned short* __restrict__ Bp,
                            const float* __restrict__ bias,
                            float* __restrict__ C,
                            int M, int K) {
  const int NC = NTILES * 16;
  const int lane = threadIdx.x & 31;
  const int wave = threadIdx.x >> 5;
  const int Mtiles = (M + 15) >> 4;
  const int gw = blockIdx.x * (blockDim.x >> 5) + wave;   // row-tile index
  if (gw >= Mtiles) return;                  // whole-wave exit: EXEC all-1 for WMMA
  const int row0 = gw << 4;

  const int lh = lane & 15;                  // 0..15
  const int hi = lane >> 4;                  // 0 or 1 (lane half)
  const int arow = row0 + lh;

  v8f acc[NTILES];
#pragma unroll
  for (int t = 0; t < NTILES; ++t) acc[t] = {};

  const unsigned short* arp0 = A + (size_t)arow * K + hi * 8;
  const unsigned short* bp0  = Bp + (size_t)lane * 16;

  if (row0 + 16 <= M) {
    // ---- full tile: wave-uniform, unconditional loads, no EXEC changes ----
    for (int kb = 0; kb < K; kb += 32) {
      v16bf a = load_a_frag(arp0 + kb);
      const unsigned short* bkp = bp0 + (size_t)(kb >> 5) * NTILES * 512;
#pragma unroll
      for (int t = 0; t < NTILES; ++t) {
        v16bf b = __builtin_bit_cast(v16bf, *(const v16us*)(bkp + (size_t)t * 512));
        acc[t] = __builtin_amdgcn_wmma_f32_16x16x32_bf16(
            false, a, false, b, (short)0, acc[t], false, false);
      }
    }
  } else {
    // ---- tail tile: per-lane row guard, zero-padded A rows ----
    const bool rowok = (arow < M);
    for (int kb = 0; kb < K; kb += 32) {
      v16bf a = {};
      if (rowok) a = load_a_frag(arp0 + kb);
      const unsigned short* bkp = bp0 + (size_t)(kb >> 5) * NTILES * 512;
#pragma unroll
      for (int t = 0; t < NTILES; ++t) {
        v16bf b = __builtin_bit_cast(v16bf, *(const v16us*)(bkp + (size_t)t * 512));
        acc[t] = __builtin_amdgcn_wmma_f32_16x16x32_bf16(
            false, a, false, b, (short)0, acc[t], false, false);
      }
    }
  }

#pragma unroll
  for (int t = 0; t < NTILES; ++t) {
    const int col = t * 16 + lh;
    const float bv = bias ? bias[col] : 0.0f;
#pragma unroll
    for (int r = 0; r < 8; ++r) {
      int row = row0 + hi * 8 + r;           // C layout: lane half selects M half
      if (row < M) C[(size_t)row * NC + col] = acc[t][r] + bv;
    }
  }
}

// ------------------------------ GAT building blocks ------------------------

__global__ void k_node_scores(const float* __restrict__ h,
                              const float* __restrict__ asv,
                              const float* __restrict__ adv,
                              float* __restrict__ es, float* __restrict__ ed,
                              int n, int C) {
  int i = blockIdx.x * blockDim.x + threadIdx.x;
  if (i >= n) return;
  const float* hp = h + (size_t)i * C;
  float s = 0.f, d = 0.f;
  for (int c = 0; c < C; ++c) { float v = hp[c]; s += v * asv[c]; d += v * adv[c]; }
  es[i] = s; ed[i] = d;
}

__global__ void k_edge_max(const int* __restrict__ src, const int* __restrict__ dst,
                           const float* __restrict__ es, const float* __restrict__ ed,
                           float* __restrict__ ebuf, float* __restrict__ m, int E) {
  int i = blockIdx.x * blockDim.x + threadIdx.x;
  if (i >= E) return;
  float e = es[src[i]] + ed[dst[i]];
  e = (e > 0.f) ? e : LEAKY_SLOPE * e;
  ebuf[i] = e;
  atomicMaxF(&m[dst[i]], e);
}

__global__ void k_self_max(const float* __restrict__ es, const float* __restrict__ ed,
                           float* __restrict__ m, int n) {
  int i = blockIdx.x * blockDim.x + threadIdx.x;
  if (i >= n) return;
  float e = es[i] + ed[i];
  e = (e > 0.f) ? e : LEAKY_SLOPE * e;
  float cur = m[i];
  m[i] = (cur > e) ? cur : e;
}

__global__ void k_fix_max(float* __restrict__ m, int n) {
  int i = blockIdx.x * blockDim.x + threadIdx.x;
  if (i >= n) return;
  float v = m[i];
  if (!__builtin_isfinite(v)) m[i] = 0.0f;   // empty segments -> 0
}

__global__ void k_edge_sum(const int* __restrict__ dst, const float* __restrict__ ebuf,
                           const float* __restrict__ m, float* __restrict__ s, int E) {
  int i = blockIdx.x * blockDim.x + threadIdx.x;
  if (i >= E) return;
  int d = dst[i];
  atomicAdd(&s[d], expf(ebuf[i] - m[d]));
}

__global__ void k_self_sum(const float* __restrict__ es, const float* __restrict__ ed,
                           const float* __restrict__ m, float* __restrict__ s, int n) {
  int i = blockIdx.x * blockDim.x + threadIdx.x;
  if (i >= n) return;
  float e = es[i] + ed[i];
  e = (e > 0.f) ? e : LEAKY_SLOPE * e;
  s[i] += expf(e - m[i]);
}

__global__ void k_edge_scatter(const int* __restrict__ src, const int* __restrict__ dst,
                               const float* __restrict__ ebuf,
                               const float* __restrict__ m, const float* __restrict__ s,
                               const float* __restrict__ h, float* __restrict__ agg,
                               float* __restrict__ alpha_out, int E, int C) {
  int i = blockIdx.x * blockDim.x + threadIdx.x;
  if (i >= E) return;
  int d = dst[i];
  float alpha = expf(ebuf[i] - m[d]) / (s[d] + 1e-16f);
  if (alpha_out) alpha_out[i] = alpha;
  const float* hp = h + (size_t)src[i] * C;
  __builtin_prefetch(hp, 0, 0);              // global_prefetch_b8
  float* ap = agg + (size_t)d * C;
  for (int c = 0; c < C; ++c) atomicAdd(&ap[c], alpha * hp[c]);
}

__global__ void k_self_scatter(const float* __restrict__ es, const float* __restrict__ ed,
                               const float* __restrict__ m, const float* __restrict__ s,
                               const float* __restrict__ h, float* __restrict__ agg,
                               int n, int C) {
  int i = blockIdx.x * blockDim.x + threadIdx.x;
  if (i >= n) return;
  float e = es[i] + ed[i];
  e = (e > 0.f) ? e : LEAKY_SLOPE * e;
  float alpha = expf(e - m[i]) / (s[i] + 1e-16f);
  const float* hp = h + (size_t)i * C;
  float* ap = agg + (size_t)i * C;
  for (int c = 0; c < C; ++c) ap[c] += alpha * hp[c];
}

__global__ void k_finalize_elu(const float* __restrict__ agg, const float* __restrict__ b,
                               unsigned short* __restrict__ actb, long total, int C) {
  long i = (long)blockIdx.x * blockDim.x + threadIdx.x;
  if (i >= total) return;
  int c = (int)(i % C);
  float v = agg[i] + b[c];
  v = (v > 0.f) ? v : (expf(v) - 1.0f);
  actb[i] = f2bf(v);
}

__global__ void k_finalize_relu_bf16(const float* __restrict__ agg, const float* __restrict__ b,
                                     unsigned short* __restrict__ outb, long total, int C) {
  long i = (long)blockIdx.x * blockDim.x + threadIdx.x;
  if (i >= total) return;
  int c = (int)(i % C);
  float v = agg[i] + b[c];
  outb[i] = f2bf(v > 0.f ? v : 0.f);
}

__global__ void k_finalize_add(const float* __restrict__ agg, const float* __restrict__ b,
                               float* __restrict__ out, long total, int C) {
  long i = (long)blockIdx.x * blockDim.x + threadIdx.x;
  if (i >= total) return;
  int c = (int)(i % C);
  out[i] = agg[i] + b[c];
}

// ------------------------------ host launcher ------------------------------

extern "C" void kernel_launch(void* const* d_in, const int* in_sizes, int n_in,
                              void* d_out, int out_size, void* d_ws, size_t ws_size,
                              hipStream_t stream) {
  const int DIN = 256, DH = 64, DOUT = 32;
  const float* x   = (const float*)d_in[0];
  const int*   ei  = (const int*)d_in[1];
  const float* W1  = (const float*)d_in[2];
  const float* a1s = (const float*)d_in[3];
  const float* a1d = (const float*)d_in[4];
  const float* b1  = (const float*)d_in[5];
  const float* W2  = (const float*)d_in[6];
  const float* a2s = (const float*)d_in[7];
  const float* a2d = (const float*)d_in[8];
  const float* b2  = (const float*)d_in[9];
  const float* W3  = (const float*)d_in[10];
  const float* a3s = (const float*)d_in[11];
  const float* a3d = (const float*)d_in[12];
  const float* b3  = (const float*)d_in[13];
  const float* Wl  = (const float*)d_in[14];
  const float* bl  = (const float*)d_in[15];

  const int N = in_sizes[0] / DIN;
  const int E = in_sizes[1] / 2;
  const int* src = ei;
  const int* dst = ei + E;

  float* out_mean  = (float*)d_out;
  float* out_var   = out_mean + (size_t)N * DOUT;
  float* out_alpha = out_var  + (size_t)N * DOUT;

  // workspace carve
  char* wsb = (char*)d_ws;
  size_t off = 0;
  auto alloc = [&](size_t bytes) -> void* {
    off = (off + 255) & ~(size_t)255;
    void* p = wsb + off;
    off += bytes;
    return p;
  };
  unsigned short* xb   = (unsigned short*)alloc((size_t)N * DIN * 2);
  unsigned short* w1p  = (unsigned short*)alloc((size_t)DIN * DH * 2);
  unsigned short* w2p  = (unsigned short*)alloc((size_t)DH * DOUT * 2);
  unsigned short* w3p  = (unsigned short*)alloc((size_t)DH * DOUT * 2);
  unsigned short* wlp  = (unsigned short*)alloc((size_t)DOUT * DOUT * 2);
  unsigned short* h1b  = (unsigned short*)alloc((size_t)N * DH * 2);
  unsigned short* mpb  = (unsigned short*)alloc((size_t)N * DOUT * 2);
  float* h1   = (float*)alloc((size_t)N * DH * 4);
  float* h2   = (float*)alloc((size_t)N * DOUT * 4);
  float* h3   = (float*)alloc((size_t)N * DOUT * 4);
  float* agg1 = (float*)alloc((size_t)N * DH * 4);
  float* agg2 = (float*)alloc((size_t)N * DOUT * 4);
  float* agg3 = (float*)alloc((size_t)N * DOUT * 4);
  float* es1  = (float*)alloc((size_t)N * 4);
  float* ed1  = (float*)alloc((size_t)N * 4);
  float* m1   = (float*)alloc((size_t)N * 4);
  float* s1   = (float*)alloc((size_t)N * 4);
  float* es2  = (float*)alloc((size_t)N * 4);
  float* ed2  = (float*)alloc((size_t)N * 4);
  float* m2   = (float*)alloc((size_t)N * 4);
  float* s2   = (float*)alloc((size_t)N * 4);
  float* es3  = (float*)alloc((size_t)N * 4);
  float* ed3  = (float*)alloc((size_t)N * 4);
  float* m3   = (float*)alloc((size_t)N * 4);
  float* s3   = (float*)alloc((size_t)N * 4);
  float* e1   = (float*)alloc((size_t)E * 4);
  float* e2   = (float*)alloc((size_t)E * 4);
  float* e3   = (float*)alloc((size_t)E * 4);

  auto blocks = [](long n) { return (unsigned)((n + TPB - 1) / TPB); };
  auto gemm_blocks = [](int M) {
    int Mtiles = (M + 15) / 16;
    return (unsigned)((Mtiles + 7) / 8);     // 8 waves (256 thr) per block
  };

  const long nE = E, nN = N;

  // --- GEMM operand prep: A row-major bf16, B packed WMMA fragments ---
  k_cvt_bf16<<<blocks((long)N * DIN), TPB, 0, stream>>>(x, xb, (long)N * DIN);
  k_pack_b<<<blocks((long)DIN * DH), TPB, 0, stream>>>(W1, w1p, DIN, DH);
  k_pack_b<<<blocks((long)DH * DOUT), TPB, 0, stream>>>(W2, w2p, DH, DOUT);
  k_pack_b<<<blocks((long)DH * DOUT), TPB, 0, stream>>>(W3, w3p, DH, DOUT);
  k_pack_b<<<blocks((long)DOUT * DOUT), TPB, 0, stream>>>(Wl, wlp, DOUT, DOUT);

  // ========================= Layer 1: GAT(x) -> h1 =========================
  k_gemm_wmma<4><<<gemm_blocks(N), TPB, 0, stream>>>(xb, w1p, nullptr, h1, N, DIN);
  k_node_scores<<<blocks(nN), TPB, 0, stream>>>(h1, a1s, a1d, es1, ed1, N, DH);
  k_init_f32<<<blocks(nN), TPB, 0, stream>>>(m1, -__builtin_inff(), nN);
  k_init_f32<<<blocks(nN), TPB, 0, stream>>>(s1, 0.0f, nN);
  k_init_f32<<<blocks((long)N * DH), TPB, 0, stream>>>(agg1, 0.0f, (long)N * DH);
  k_edge_max<<<blocks(nE), TPB, 0, stream>>>(src, dst, es1, ed1, e1, m1, E);
  k_fix_max<<<blocks(nN), TPB, 0, stream>>>(m1, N);
  k_edge_sum<<<blocks(nE), TPB, 0, stream>>>(dst, e1, m1, s1, E);
  k_edge_scatter<<<blocks(nE), TPB, 0, stream>>>(src, dst, e1, m1, s1, h1, agg1, nullptr, E, DH);
  k_finalize_elu<<<blocks((long)N * DH), TPB, 0, stream>>>(agg1, b1, h1b, (long)N * DH, DH);

  // =================== Layer 2: GAT(h1) -> mean, alpha =====================
  k_gemm_wmma<2><<<gemm_blocks(N), TPB, 0, stream>>>(h1b, w2p, nullptr, h2, N, DH);
  k_node_scores<<<blocks(nN), TPB, 0, stream>>>(h2, a2s, a2d, es2, ed2, N, DOUT);
  k_init_f32<<<blocks(nN), TPB, 0, stream>>>(m2, -__builtin_inff(), nN);
  k_init_f32<<<blocks(nN), TPB, 0, stream>>>(s2, 0.0f, nN);
  k_init_f32<<<blocks((long)N * DOUT), TPB, 0, stream>>>(agg2, 0.0f, (long)N * DOUT);
  k_edge_max<<<blocks(nE), TPB, 0, stream>>>(src, dst, es2, ed2, e2, m2, E);
  k_fix_max<<<blocks(nN), TPB, 0, stream>>>(m2, N);
  k_edge_sum<<<blocks(nE), TPB, 0, stream>>>(dst, e2, m2, s2, E);
  k_edge_scatter<<<blocks(nE), TPB, 0, stream>>>(src, dst, e2, m2, s2, h2, agg2, out_alpha, E, DOUT);
  k_finalize_relu_bf16<<<blocks((long)N * DOUT), TPB, 0, stream>>>(agg2, b2, mpb, (long)N * DOUT, DOUT);
  // mean = relu(...) @ Wl + bl
  k_gemm_wmma<2><<<gemm_blocks(N), TPB, 0, stream>>>(mpb, wlp, bl, out_mean, N, DOUT);

  // ============ Layer 3 (variance head, with self-loops) -> var ============
  k_gemm_wmma<2><<<gemm_blocks(N), TPB, 0, stream>>>(h1b, w3p, nullptr, h3, N, DH);
  k_node_scores<<<blocks(nN), TPB, 0, stream>>>(h3, a3s, a3d, es3, ed3, N, DOUT);
  k_init_f32<<<blocks(nN), TPB, 0, stream>>>(m3, -__builtin_inff(), nN);
  k_init_f32<<<blocks(nN), TPB, 0, stream>>>(s3, 0.0f, nN);
  k_init_f32<<<blocks((long)N * DOUT), TPB, 0, stream>>>(agg3, 0.0f, (long)N * DOUT);
  k_edge_max<<<blocks(nE), TPB, 0, stream>>>(src, dst, es3, ed3, e3, m3, E);
  k_self_max<<<blocks(nN), TPB, 0, stream>>>(es3, ed3, m3, N);
  k_fix_max<<<blocks(nN), TPB, 0, stream>>>(m3, N);
  k_edge_sum<<<blocks(nE), TPB, 0, stream>>>(dst, e3, m3, s3, E);
  k_self_sum<<<blocks(nN), TPB, 0, stream>>>(es3, ed3, m3, s3, N);
  k_edge_scatter<<<blocks(nE), TPB, 0, stream>>>(src, dst, e3, m3, s3, h3, agg3, nullptr, E, DOUT);
  k_self_scatter<<<blocks(nN), TPB, 0, stream>>>(es3, ed3, m3, s3, h3, agg3, N, DOUT);
  k_finalize_add<<<blocks((long)N * DOUT), TPB, 0, stream>>>(agg3, b3, out_var, (long)N * DOUT, DOUT);

  (void)n_in; (void)out_size; (void)ws_size;
}
